// TransformerDecoderUnit_89764816486648
// MI455X (gfx1250) — matrix operand order repaired and checked by reading, not compiled, not run on hardware
//
#include <hip/hip_runtime.h>
#include <cmath>

#define NTOK 2304          // 48*48 tokens
#define FEATC 256
#define NB 4
#define NHEAD 8
#define CPAD 288           // 286 -> padded to multiple of 32
#define HW 48
#define OBLK 4             // o-tiles per wave in GEMM (16x64 output strip)

typedef __attribute__((ext_vector_type(16))) __bf16          v16bf;
typedef __attribute__((ext_vector_type(16))) unsigned short  v16us;
typedef __attribute__((ext_vector_type(8)))  unsigned short  v8us;
typedef __attribute__((ext_vector_type(8)))  float           v8f;

__device__ __forceinline__ unsigned short f2bf(float f) {
  union { float f; unsigned u; } x; x.f = f;
  unsigned u = x.u;
  unsigned r = u + 0x7fffu + ((u >> 16) & 1u);   // round-to-nearest-even
  return (unsigned short)(r >> 16);
}
__device__ __forceinline__ v16bf asbf(v16us u) { return __builtin_bit_cast(v16bf, u); }
__device__ __forceinline__ v16us cat8(v8us a, v8us b) {
  v16us r;
  #pragma unroll
  for (int j = 0; j < 8; ++j) { r[j] = a[j]; r[8 + j] = b[j]; }
  return r;
}
__device__ __forceinline__ v8f wmma_bf(v16bf a, v16bf b, v8f c) {
  // D(16x16 f32) = A(16x32 bf16) x B(32x16 bf16) + C
  return __builtin_amdgcn_wmma_f32_16x16x32_bf16(false, a, false, b, (short)0, c, false, false);
}

// ---------------- positional encoding: pe[30][2304] f32 ----------------
__global__ void pe_kernel(float* __restrict__ pe) {
  int idx = blockIdx.x * blockDim.x + threadIdx.x;
  if (idx >= 30 * NTOK) return;
  int ch = idx / NTOK, n = idx % NTOK;
  int y = n / HW, x = n % HW;
  int g = ch / 10, i = ch % 10;
  const float SCALE = 6.28318530717958647692f;
  float e;
  if (g == 0)      e = ((float)(y + 1) / (HW + 1e-6f)) * SCALE;
  else if (g == 1) e = ((float)(x + 1) / (HW + 1e-6f)) * SCALE;
  else             e = SCALE;
  float t = powf(10000.0f, (float)(2 * (i / 2)) / 10.0f);
  float p = e / t;
  pe[idx] = (i & 1) ? cosf(p) : sinf(p);
}

// ---- build token-major bf16 concat inputs: qct/kct [b][n][288] ----
__global__ void packqk_kernel(const float* __restrict__ q, const float* __restrict__ k,
                              const float* __restrict__ pe,
                              unsigned short* __restrict__ qct, unsigned short* __restrict__ kct) {
  int idx = blockIdx.x * blockDim.x + threadIdx.x;      // NB*NTOK*CPAD
  if (idx >= NB * NTOK * CPAD) return;
  int b  = idx / (NTOK * CPAD);
  int r  = idx - b * NTOK * CPAD;
  int n  = r / CPAD, ch = r % CPAD;
  unsigned short qv, kv;
  if (ch < FEATC) {
    size_t s = ((size_t)b * FEATC + ch) * NTOK + n;
    qv = f2bf(q[s]); kv = f2bf(k[s]);
  } else if (ch < FEATC + 30) {
    unsigned short p = f2bf(pe[(size_t)(ch - FEATC) * NTOK + n]);
    qv = p; kv = p;
  } else { qv = 0; kv = 0; }
  qct[idx] = qv; kct[idx] = kv;
}

// ---- build token-major bf16 v: vt [b][n][256] ----
__global__ void packv_kernel(const float* __restrict__ v, unsigned short* __restrict__ vt) {
  int idx = blockIdx.x * blockDim.x + threadIdx.x;      // NB*NTOK*FEATC
  if (idx >= NB * NTOK * FEATC) return;
  int b  = idx / (NTOK * FEATC);
  int r  = idx - b * NTOK * FEATC;
  int n  = r / FEATC, ch = r % FEATC;
  vt[idx] = f2bf(v[((size_t)b * FEATC + ch) * NTOK + n]);
}

// ---- convert (and zero-pad) weight matrix f32 [O][Cin] -> bf16 [O][Cpad] ----
__global__ void wcvt_kernel(const float* __restrict__ src, unsigned short* __restrict__ dst,
                            int O, int Cin, int Cpad) {
  int idx = blockIdx.x * blockDim.x + threadIdx.x;
  if (idx >= O * Cpad) return;
  int o = idx / Cpad, c = idx % Cpad;
  dst[idx] = (c < Cin) ? f2bf(src[(size_t)o * Cin + c]) : (unsigned short)0;
}

// ---------------- token-major bf16 WMMA GEMM, 16x64 strip per wave ----------------
// Yt[b][n][o] = sum_c Xt[b][n][c] * Wb[o][c]      (D = A(tokens x C) * B(C x o))
// A tile loaded once per c-step, reused by OBLK=4 WMMAs (4 independent accumulators).
// epilogue: *scale, +bias, relu, +resid (f32 [b][n][o]);
// outf: f32 [b][n][o] (optional); outb: bf16, vmode0 [b][n][o] / vmode1 [b][o][n] (optional)
__global__ void gemm_tm_kernel(const unsigned short* __restrict__ A,
                               const unsigned short* __restrict__ Wb,
                               const float* __restrict__ bias, const float* __restrict__ resid,
                               float* __restrict__ outf, unsigned short* __restrict__ outb,
                               int C, int OC, int nB, float scale, int relu, int vmode) {
  int wave = (int)((blockIdx.x * blockDim.x + threadIdx.x) >> 5);
  int lane = threadIdx.x & 31;
  int lg = lane & 15, hi = lane >> 4;
  int os = OC >> 6;                         // 64-wide o-strips
  const int nt = NTOK >> 4;                 // 144 token tiles
  int total = nB * nt * os;
  if (wave >= total) return;                // wave-uniform
  int b = wave / (nt * os);
  int rem = wave - b * nt * os;
  int ni = rem / os, oi = rem - ni * os;
  int nbase = ni << 4, obase = oi << 6;

  v8f acc[OBLK];
  #pragma unroll
  for (int t = 0; t < OBLK; ++t) acc[t] = (v8f){};

  const unsigned short* arowb = A + ((size_t)b * NTOK + nbase + lg) * C;
  const unsigned short* wrowb = Wb + (size_t)(obase + lg) * C;
  const size_t wtstep = (size_t)16 * C;     // next o-tile's row block
  for (int c = 0; c < C; c += 32) {
    // A: 16 tokens x 32 channels, contiguous k = 16*hi + s: one aligned 32B load, reused OBLK times
    v16us au = *(const v16us*)(arowb + c + hi * 16);
    v16bf av = asbf(au);
    #pragma unroll
    for (int t = 0; t < OBLK; ++t) {
      // B: 32 channels x 16 outputs (weight row = column): one aligned 32B load
      v16us bu = *(const v16us*)(wrowb + (size_t)t * wtstep + c + hi * 16);
      acc[t] = wmma_bf(av, asbf(bu), acc[t]);
    }
  }

  #pragma unroll
  for (int t = 0; t < OBLK; ++t) {
    int o = obase + t * 16 + lg;            // D layout: N = lane (within 16-group)
    float bv = bias ? bias[o] : 0.f;
    #pragma unroll
    for (int r = 0; r < 8; ++r) {
      int token = nbase + hi * 8 + r;       // D layout: M = r (+8 for hi lanes)
      size_t tmIdx = ((size_t)b * NTOK + token) * OC + o;
      float y = acc[t][r] * scale + bv;
      if (relu)  y = fmaxf(y, 0.f);
      if (resid) y += resid[tmIdx];
      if (outf)  outf[tmIdx] = y;
      if (outb) {
        size_t oi2 = vmode ? ((size_t)(b * OC + o)) * NTOK + token : tmIdx;
        outb[oi2] = f2bf(y);
      }
    }
  }
}

// ---------------- flash attention: one wave per (b, head, 16-query tile) ----------------
// Qt,Kt: bf16 token-major [b][n][256] (o = h*32+d)   Vb: bf16 [b][h*32+dv][n]
// Ot: bf16 token-major [b][n][256]
__global__ void attn_kernel(const unsigned short* __restrict__ Qt,
                            const unsigned short* __restrict__ Kt,
                            const unsigned short* __restrict__ Vb,
                            unsigned short* __restrict__ Ot) {
  __shared__ float Pst[8 * 16 * 36];        // per-wave 16x32 P tile, stride 36: bank-conflict-free
  int wave = (int)((blockIdx.x * blockDim.x + threadIdx.x) >> 5);
  int lane = threadIdx.x & 31;
  int lg = lane & 15, hi = lane >> 4;
  const int QT = NTOK / 16;                 // 144
  int qt = wave % QT;
  int bh = wave / QT;
  if (bh >= NB * NHEAD) return;
  int b = bh >> 3, h = bh & 7;
  float* P = &Pst[(threadIdx.x >> 5) * (16 * 36)];

  // Q tile 16x32 (A layout); 1/sqrt(dk) folded into Q projection
  const unsigned short* qrow = Qt + ((size_t)b * NTOK + qt * 16 + lg) * FEATC + h * 32;
  v16us qu = *(const v16us*)(qrow + hi * 16);
  v16bf qa = asbf(qu);

  v8f acc0 = {}, acc1 = {};
  float mrow[8], lrow[8];
  #pragma unroll
  for (int r = 0; r < 8; ++r) { mrow[r] = -1e30f; lrow[r] = 0.f; }

  for (int kp = 0; kp < NTOK / 32; ++kp) {  // 72 steps of 32 keys
    int k0 = kp * 32;
    // B = K^T tiles: lane = key column, elems = d (one aligned 32B load each)
    const unsigned short* kr0 = Kt + ((size_t)b * NTOK + k0 + lg) * FEATC + h * 32 + hi * 16;
    const unsigned short* kr1 = Kt + ((size_t)b * NTOK + k0 + 16 + lg) * FEATC + h * 32 + hi * 16;
    v16us ku0 = *(const v16us*)kr0;
    v16us ku1 = *(const v16us*)kr1;
    v8f zero = {};
    v8f s0 = wmma_bf(qa, asbf(ku0), zero);  // S[q][k0+0..15]
    v8f s1 = wmma_bf(qa, asbf(ku1), zero);  // S[q][k0+16..31]

    // online softmax: stats live exactly in D-layout element rows
    float mnew[8], alpha[8];
    #pragma unroll
    for (int r = 0; r < 8; ++r) {
      float v = fmaxf(s0[r], s1[r]);
      v = fmaxf(v, __shfl_xor(v, 1));
      v = fmaxf(v, __shfl_xor(v, 2));
      v = fmaxf(v, __shfl_xor(v, 4));
      v = fmaxf(v, __shfl_xor(v, 8));
      mnew[r] = fmaxf(mrow[r], v);
      alpha[r] = __expf(mrow[r] - mnew[r]);
      mrow[r] = mnew[r];
    }
    #pragma unroll
    for (int r = 0; r < 8; ++r) {
      float p0 = __expf(s0[r] - mnew[r]);
      float p1 = __expf(s1[r] - mnew[r]);
      float ps = p0 + p1;
      ps += __shfl_xor(ps, 1);
      ps += __shfl_xor(ps, 2);
      ps += __shfl_xor(ps, 4);
      ps += __shfl_xor(ps, 8);
      lrow[r] = lrow[r] * alpha[r] + ps;
      acc0[r] *= alpha[r];
      acc1[r] *= alpha[r];
      P[(hi * 8 + r) * 36 + lg]      = p0;  // D layout -> LDS [row][key]
      P[(hi * 8 + r) * 36 + 16 + lg] = p1;
    }
    // read back in A layout, cvt to bf16
    v16us pu;
    #pragma unroll
    for (int j = 0; j < 8; ++j) {
      pu[j]     = f2bf(P[lg * 36 + hi * 16 + j]);
      pu[8 + j] = f2bf(P[lg * 36 + hi * 16 + 8 + j]);
    }
    v16bf pa = asbf(pu);
    // B = V tiles (dv-major): lane = dv column, elems = keys (aligned 32B loads)
    const unsigned short* vr0 = Vb + ((size_t)bh * 32 + lg) * NTOK + k0 + hi * 16;
    const unsigned short* vr1 = Vb + ((size_t)bh * 32 + 16 + lg) * NTOK + k0 + hi * 16;
    v16us vu0 = *(const v16us*)vr0;
    v16us vu1 = *(const v16us*)vr1;
    acc0 = wmma_bf(pa, asbf(vu0), acc0);    // O[q][dv 0..15]
    acc1 = wmma_bf(pa, asbf(vu1), acc1);    // O[q][dv 16..31]
  }

  #pragma unroll
  for (int r = 0; r < 8; ++r) {
    float inv = 1.0f / lrow[r];
    int token = qt * 16 + hi * 8 + r;
    size_t base = ((size_t)b * NTOK + token) * FEATC + h * 32;
    Ot[base + lg]      = f2bf(acc0[r] * inv);
    Ot[base + 16 + lg] = f2bf(acc1[r] * inv);
  }
}

// ---------------- BatchNorm over (b,n) per channel; transposes to [b][c][n] ----------------
__global__ void bn_kernel(const float* __restrict__ x /*[b][n][256]*/,
                          const float* __restrict__ gamma, const float* __restrict__ beta,
                          float* __restrict__ out /*[b][256][n]*/) {
  int c = blockIdx.x;                       // 256 blocks
  __shared__ float ssum[256], ssq[256];
  float s = 0.f, s2 = 0.f;
  for (int i = threadIdx.x; i < NB * NTOK; i += blockDim.x) {
    float v = x[(size_t)i * FEATC + c];
    s += v; s2 += v * v;
  }
  ssum[threadIdx.x] = s; ssq[threadIdx.x] = s2;
  __syncthreads();
  for (int st = 128; st > 0; st >>= 1) {
    if ((int)threadIdx.x < st) { ssum[threadIdx.x] += ssum[threadIdx.x + st];
                                 ssq[threadIdx.x]  += ssq[threadIdx.x + st]; }
    __syncthreads();
  }
  float mean = ssum[0] / (float)(NB * NTOK);
  float var  = ssq[0] / (float)(NB * NTOK) - mean * mean;
  float rstd = rsqrtf(var + 1e-5f);
  float ga = gamma[c], be = beta[c];
  for (int i = threadIdx.x; i < NB * NTOK; i += blockDim.x) {
    int b = i / NTOK, n = i % NTOK;
    float v = x[(size_t)i * FEATC + c];
    out[((size_t)b * FEATC + c) * NTOK + n] = (v - mean) * rstd * ga + be;
  }
}

extern "C" void kernel_launch(void* const* d_in, const int* in_sizes, int n_in,
                              void* d_out, int out_size, void* d_ws, size_t ws_size,
                              hipStream_t stream) {
  const float* q     = (const float*)d_in[0];
  const float* k     = (const float*)d_in[1];
  const float* v     = (const float*)d_in[2];
  const float* wq    = (const float*)d_in[3];
  const float* wk    = (const float*)d_in[4];
  const float* wv    = (const float*)d_in[5];
  const float* wfc   = (const float*)d_in[6];
  const float* w1    = (const float*)d_in[7];
  const float* b1    = (const float*)d_in[8];
  const float* w2    = (const float*)d_in[9];
  const float* b2    = (const float*)d_in[10];
  const float* gamma = (const float*)d_in[11];
  const float* beta  = (const float*)d_in[12];
  float* out = (float*)d_out;
  char* ws = (char*)d_ws;

  // workspace layout (bytes); peak ~35.3 MB with aliasing
  const size_t o_pe   = 0;
  const size_t o_qct  = o_pe   + (size_t)30 * NTOK * 4;
  const size_t o_kct  = o_qct  + (size_t)NB * NTOK * CPAD * 2;
  const size_t o_vt   = o_kct  + (size_t)NB * NTOK * CPAD * 2;
  const size_t o_wqb  = o_vt   + (size_t)NB * NTOK * FEATC * 2;
  const size_t o_wkb  = o_wqb  + (size_t)FEATC * CPAD * 2;
  const size_t o_wvb  = o_wkb  + (size_t)FEATC * CPAD * 2;
  const size_t o_wfcb = o_wvb  + (size_t)FEATC * FEATC * 2;
  const size_t o_w1b  = o_wfcb + (size_t)FEATC * FEATC * 2;
  const size_t o_w2b  = o_w1b  + (size_t)FEATC * FEATC * 2;
  const size_t o_Qt   = o_w2b  + (size_t)FEATC * FEATC * 2;
  const size_t o_Kt   = o_Qt   + (size_t)NB * NTOK * FEATC * 2;
  const size_t o_Vb   = o_Kt   + (size_t)NB * NTOK * FEATC * 2;
  const size_t o_Ot   = o_Vb   + (size_t)NB * NTOK * FEATC * 2;
  const size_t need   = o_Ot   + (size_t)NB * NTOK * FEATC * 2;
  if (ws_size < need) return;
  // aliases (sources dead before writes)
  const size_t o_O2b = o_Qt;    // Qt dead after attention
  const size_t o_O2f = o_Kt;    // Kt+Vb dead after attention (f32 needs both regions)
  const size_t o_h1b = o_vt;    // vt dead after V projection
  const size_t o_pre = o_qct;   // qct(+kct head) dead after Q/K projections

  float* pe            = (float*)(ws + o_pe);
  unsigned short* qct  = (unsigned short*)(ws + o_qct);
  unsigned short* kct  = (unsigned short*)(ws + o_kct);
  unsigned short* vt   = (unsigned short*)(ws + o_vt);
  unsigned short* wqb  = (unsigned short*)(ws + o_wqb);
  unsigned short* wkb  = (unsigned short*)(ws + o_wkb);
  unsigned short* wvb  = (unsigned short*)(ws + o_wvb);
  unsigned short* wfcb = (unsigned short*)(ws + o_wfcb);
  unsigned short* w1b  = (unsigned short*)(ws + o_w1b);
  unsigned short* w2b  = (unsigned short*)(ws + o_w2b);
  unsigned short* Qt   = (unsigned short*)(ws + o_Qt);
  unsigned short* Kt   = (unsigned short*)(ws + o_Kt);
  unsigned short* Vb   = (unsigned short*)(ws + o_Vb);
  unsigned short* Ot   = (unsigned short*)(ws + o_Ot);
  unsigned short* O2b  = (unsigned short*)(ws + o_O2b);
  float*          O2f  = (float*)(ws + o_O2f);
  unsigned short* h1b  = (unsigned short*)(ws + o_h1b);
  float*          pre  = (float*)(ws + o_pre);

  const float qscale = 0.17677669529663688f;     // 1/sqrt(32)

  // ---- prep: pe, pack inputs to token-major bf16, convert weights ----
  pe_kernel<<<(30 * NTOK + 255) / 256, 256, 0, stream>>>(pe);
  packqk_kernel<<<(NB * NTOK * CPAD) / 256, 256, 0, stream>>>(q, k, pe, qct, kct);
  packv_kernel<<<(NB * NTOK * FEATC) / 256, 256, 0, stream>>>(v, vt);
  wcvt_kernel<<<(FEATC * CPAD) / 256, 256, 0, stream>>>(wq, wqb, FEATC, 286, CPAD);
  wcvt_kernel<<<(FEATC * CPAD) / 256, 256, 0, stream>>>(wk, wkb, FEATC, 286, CPAD);
  wcvt_kernel<<<(FEATC * FEATC) / 256, 256, 0, stream>>>(wv, wvb, FEATC, FEATC, FEATC);
  wcvt_kernel<<<(FEATC * FEATC) / 256, 256, 0, stream>>>(wfc, wfcb, FEATC, FEATC, FEATC);
  wcvt_kernel<<<(FEATC * FEATC) / 256, 256, 0, stream>>>(w1, w1b, FEATC, FEATC, FEATC);
  wcvt_kernel<<<(FEATC * FEATC) / 256, 256, 0, stream>>>(w2, w2b, FEATC, FEATC, FEATC);

  // ---- projections (WMMA, token-major, 16x64 strips): 2304 waves -> 288 blocks ----
  const int gemmBlocks = (NB * (NTOK / 16) * (FEATC / 64) * 32 + 255) / 256;
  gemm_tm_kernel<<<gemmBlocks, 256, 0, stream>>>(qct, wqb, nullptr, nullptr, nullptr, Qt,
      CPAD, FEATC, NB, qscale, 0, 0);
  gemm_tm_kernel<<<gemmBlocks, 256, 0, stream>>>(kct, wkb, nullptr, nullptr, nullptr, Kt,
      CPAD, FEATC, NB, 1.0f, 0, 0);
  gemm_tm_kernel<<<gemmBlocks, 256, 0, stream>>>(vt, wvb, nullptr, nullptr, nullptr, Vb,
      FEATC, FEATC, NB, 1.0f, 0, 1);                     // dv-major for attention B-operand

  // ---- attention: 4608 waves -> 576 blocks ----
  attn_kernel<<<576, 256, 0, stream>>>(Qt, Kt, Vb, Ot);

  // ---- wfc + FFN + residual (WMMA chain, bf16 activations) ----
  gemm_tm_kernel<<<gemmBlocks, 256, 0, stream>>>(Ot, wfcb, nullptr, nullptr, O2f, O2b,
      FEATC, FEATC, NB, 1.0f, 0, 0);
  gemm_tm_kernel<<<gemmBlocks, 256, 0, stream>>>(O2b, w1b, b1, nullptr, nullptr, h1b,
      FEATC, FEATC, NB, 1.0f, 1, 0);
  gemm_tm_kernel<<<gemmBlocks, 256, 0, stream>>>(h1b, w2b, b2, O2f, pre, nullptr,
      FEATC, FEATC, NB, 1.0f, 0, 0);

  // ---- batchnorm (+ transpose back to [b][c][h][w]) ----
  bn_kernel<<<FEATC, 256, 0, stream>>>(pre, gamma, beta, out);
}
